// GaussianMixtureAdaptiveProposal_12979391169164
// MI455X (gfx1250) — matrix operand order repaired and checked by reading, not compiled, not run on hardware
//
#include <hip/hip_runtime.h>
#include <hip/hip_bf16.h>

typedef __attribute__((ext_vector_type(16))) _Float16 v16h;
typedef __attribute__((ext_vector_type(8)))  float    v8f;

#define NN 8192
#define MM 4096
#define DD 784
#define KP 800           // D padded to multiple of 32
#define KT 25            // KP/32
#define NTILES (NN/16)   // 512
#define MTILES (MM/16)   // 256
#define LOG_2PI 1.8378770664093453f
#define LOG_M   8.317766166719343f   // log(4096)

// ---------------------------------------------------------------------------
// Prep: inv_std + log-const
// ---------------------------------------------------------------------------
__global__ void k_std(const float* __restrict__ stdv, float* __restrict__ inv_std,
                      float* __restrict__ logconst) {
    __shared__ float red[256];
    int t = threadIdx.x;
    float acc = 0.f;
    for (int k = t; k < DD; k += 256) {
        float s = stdv[k];
        inv_std[k] = 1.0f / s;
        acc += __logf(s);
    }
    red[t] = acc; __syncthreads();
    for (int off = 128; off > 0; off >>= 1) {
        if (t < off) red[t] += red[t + off];
        __syncthreads();
    }
    if (t == 0) *logconst = -0.5f * (float)DD * LOG_2PI - red[0];
}

// ---------------------------------------------------------------------------
// Prep: per-row squared norm of scaled vectors (one block per row)
// ---------------------------------------------------------------------------
__global__ void k_norm(const float* __restrict__ src, const float* __restrict__ inv_std,
                       float* __restrict__ dst) {
    __shared__ float red[256];
    int n = blockIdx.x, t = threadIdx.x;
    const float* p = src + (size_t)n * DD;
    float acc = 0.f;
    for (int k = t; k < DD; k += 256) {
        float u = p[k] * inv_std[k];
        acc += u * u;
    }
    red[t] = acc; __syncthreads();
    for (int off = 128; off > 0; off >>= 1) {
        if (t < off) red[t] += red[t + off];
        __syncthreads();
    }
    if (t == 0) dst[n] = red[0];
}

// ---------------------------------------------------------------------------
// Prep: pack samples -> A-fragment layout, split f16 hi/lo.
// Global layout: per 32-row strip: [ hi: [ntL(2)][kt(25)][lane(32)][j(16)] | lo: same ]
// A 16x32 f16 fragment (ISA 7.12.2): lanes 0-15: K {0-7,16-23}; lanes 16-31: K {8-15,24-31}
// ---------------------------------------------------------------------------
__global__ void k_pack_a(const float* __restrict__ src, const float* __restrict__ inv_std,
                         _Float16* __restrict__ Apk) {
    size_t idx = (size_t)blockIdx.x * 256 + threadIdx.x;
    const size_t total = (size_t)NTILES * KT * 512;
    if (idx >= total) return;
    int j    = (int)(idx & 15);
    int lane = (int)((idx >> 4) & 31);
    int kt   = (int)((idx >> 9) % KT);
    int nt   = (int)((idx >> 9) / KT);
    int n    = nt * 16 + (lane & 15);
    int koff = ((lane >> 4) << 3) + j + ((j >> 3) << 3);
    int k    = kt * 32 + koff;
    float u = 0.f;
    if (k < DD) u = src[(size_t)n * DD + k] * inv_std[k];
    _Float16 h = (_Float16)u;
    _Float16 l = (_Float16)(u - (float)h);
    size_t base = (size_t)(nt >> 1) * 51200;                    // f16 per strip
    int    off  = (((nt & 1) * KT + kt) << 9) + (lane << 4) + j;
    Apk[base + off]         = h;
    Apk[base + 25600 + off] = l;
}

// ---------------------------------------------------------------------------
// Prep: pack x -> B-fragment layout, split f16 hi/lo.
// Global layout: [kt(25)][mt(256)][ hi 512 f16 | lo 512 f16 ]
// B 32x16 f16 fragment: lanes 0-15: K {0-7,16-23}; lanes 16-31: K {8-15,24-31} (2 per VGPR)
// ---------------------------------------------------------------------------
__global__ void k_pack_b(const float* __restrict__ src, const float* __restrict__ inv_std,
                         _Float16* __restrict__ Bpk) {
    size_t idx = (size_t)blockIdx.x * 256 + threadIdx.x;
    const size_t total = (size_t)KT * MTILES * 512;
    if (idx >= total) return;
    int j    = (int)(idx & 15);
    int lane = (int)((idx >> 4) & 31);
    int mt   = (int)((idx >> 9) % MTILES);
    int kt   = (int)((idx >> 9) / MTILES);
    int m    = mt * 16 + (lane & 15);
    int koff = ((lane >> 4) << 3) + j + ((j >> 3) << 3);
    int k    = kt * 32 + koff;
    float u = 0.f;
    if (k < DD) u = src[(size_t)m * DD + k] * inv_std[k];
    _Float16 h = (_Float16)u;
    _Float16 l = (_Float16)(u - (float)h);
    size_t d = (size_t)(kt * MTILES + mt) * 1024 + (lane << 4) + j;
    Bpk[d]       = h;
    Bpk[d + 512] = l;
}

// ---------------------------------------------------------------------------
// Fused split-f16 WMMA GEMM + online logsumexp, async-to-LDS staged.
// Grid: 256 blocks (32-row strips). Block: 256 threads = 8 waves, 2x4 tile grid.
// LDS: A strip resident 102400 B + B double-buffered chunks 2 x 40960 B = 184320 B.
// B pipeline: 5 chunks of 5 K-tiles per 64-column panel; chunk s+2 DMA overlaps compute of chunk s+1;
// per-wave chunk = 10 async b128 -> s_wait_asynccnt 0xA.
// ---------------------------------------------------------------------------
__global__ __launch_bounds__(256) void k_gemm_lse(
    const char* __restrict__ Apk, const char* __restrict__ Bpk,
    const float* __restrict__ unorm, const float* __restrict__ vnorm,
    const float* __restrict__ cptr, float* __restrict__ out) {

    extern __shared__ char smem[];
    _Float16* lsA  = (_Float16*)smem;            // [hi 25600 | lo 25600] f16
    _Float16* lsB  = (_Float16*)(smem + 102400); // 2 buffers x 20480 f16
    float*    lsRed = (float*)smem;              // reused after GEMM
    const unsigned ldsBase = (unsigned)(unsigned long long)smem;

    const int tid  = threadIdx.x;
    const int lane = tid & 31;
    const int wave = tid >> 5;
    const int wr   = wave >> 2;   // 0..1 : row tile
    const int wc   = wave & 3;    // 0..3 : col tile
    const int strip = blockIdx.x;

    // --- async DMA: resident A strip (contiguous 102400 B), 25 b128 per thread
    {
        const unsigned abase = (unsigned)strip * 102400u;
#pragma unroll
        for (int p = 0; p < 25; ++p) {
            const unsigned b = (unsigned)((tid + (p << 8)) << 4);
            asm volatile("global_load_async_to_lds_b128 %0, %1, %2"
                         :: "v"(ldsBase + b), "v"(abase + b), "s"(Apk)
                         : "memory");
        }
    }

    // --- async DMA issue for one B chunk (10 b128 per thread / per wave)
    auto issueB = [&](int s2) {
        const int mt2 = s2 / 5, c2 = s2 % 5;
        const unsigned lb = ldsBase + 102400u + (unsigned)(s2 & 1) * 40960u;
#pragma unroll
        for (int p = 0; p < 10; ++p) {
            const int i   = tid + (p << 8);          // 0..2559 uint4 of chunk
            const int ktl = i >> 9, rem = i & 511;
            const unsigned go = (unsigned)((((c2 * 5 + ktl) << 15) + (mt2 << 9) + rem) << 4);
            asm volatile("global_load_async_to_lds_b128 %0, %1, %2"
                         :: "v"(lb + (unsigned)(i << 4)), "v"(go), "s"(Bpk)
                         : "memory");
        }
    };
    issueB(0);
    issueB(1);

    // per-lane constants: 0.5*||u||^2 for the 8 rows this lane accumulates
    const int rbase = strip * 32 + wr * 16 + ((lane >> 4) << 3);
    float rn[8];
#pragma unroll
    for (int r = 0; r < 8; ++r) rn[r] = 0.5f * unorm[rbase + r];
    const float logc = *cptr;

    float lm[8], ls[8];
#pragma unroll
    for (int r = 0; r < 8; ++r) { lm[r] = -__builtin_inff(); ls[r] = 0.f; }

    for (int mt = 0; mt < MM / 64; ++mt) {
        v8f acc = {};
#pragma unroll 1
        for (int c = 0; c < 5; ++c) {
            const int s = mt * 5 + c;
            // own DMAs for chunk s (and A strip, in-order) complete:
            if (s == 319) asm volatile("s_wait_asynccnt 0x0" ::: "memory");
            else          asm volatile("s_wait_asynccnt 0xA" ::: "memory");
            __syncthreads();  // everyone's chunk-s data visible

            const _Float16* bb = lsB + (s & 1) * 20480;
#pragma unroll
            for (int ktl = 0; ktl < 5; ++ktl) {
                const int kt = c * 5 + ktl;
                v16h ah = *(const v16h*)(lsA + ((wr * KT + kt) << 9) + (lane << 4));
                v16h al = *(const v16h*)(lsA + 25600 + ((wr * KT + kt) << 9) + (lane << 4));
                v16h bh = *(const v16h*)(bb + (ktl << 12) + (wc << 10) + (lane << 4));
                v16h bl = *(const v16h*)(bb + (ktl << 12) + (wc << 10) + 512 + (lane << 4));
                acc = __builtin_amdgcn_wmma_f32_16x16x32_f16(false, ah, false, bh, (short)0, acc, false, false);
                acc = __builtin_amdgcn_wmma_f32_16x16x32_f16(false, ah, false, bl, (short)0, acc, false, false);
                acc = __builtin_amdgcn_wmma_f32_16x16x32_f16(false, al, false, bh, (short)0, acc, false, false);
            }
            __syncthreads();                  // chunk-s buffer free for reuse
            if (s + 2 < 320) issueB(s + 2);   // overlap DMA with next compute
        }

        // epilogue: logp = dot - 0.5||u||^2 - 0.5||v||^2 + logconst; online logsumexp
        const int col = mt * 64 + wc * 16 + (lane & 15);
        const float hv = 0.5f * vnorm[col];
#pragma unroll
        for (int r = 0; r < 8; ++r) {
            float v  = acc[r] - rn[r] - hv + logc;
            float nm = fmaxf(lm[r], v);
            ls[r] = ls[r] * __expf(lm[r] - nm) + __expf(v - nm);
            lm[r] = nm;
        }
    }

    // reduce the 16 column-slots within each 16-lane group
#pragma unroll
    for (int r = 0; r < 8; ++r) {
        for (int s = 1; s < 16; s <<= 1) {
            float om = __shfl_xor(lm[r], s, 32);
            float os = __shfl_xor(ls[r], s, 32);
            float nm = fmaxf(lm[r], om);
            ls[r] = ls[r] * __expf(lm[r] - nm) + os * __expf(om - nm);
            lm[r] = nm;
        }
    }

    __syncthreads();  // done reading LDS A/B; safe to reuse as reduction buffer
    if ((lane & 15) == 0) {
        int rowLoc = wr * 16 + ((lane >> 4) << 3);
#pragma unroll
        for (int r = 0; r < 8; ++r) {
            lsRed[((rowLoc + r) * 4 + wc) * 2 + 0] = lm[r];
            lsRed[((rowLoc + r) * 4 + wc) * 2 + 1] = ls[r];
        }
    }
    __syncthreads();
    if (tid < 32) {
        float m = -__builtin_inff(), s = 0.f;
#pragma unroll
        for (int c = 0; c < 4; ++c) {
            float om = lsRed[(tid * 4 + c) * 2 + 0];
            float os = lsRed[(tid * 4 + c) * 2 + 1];
            float nm = fmaxf(m, om);
            s = s * __expf(m - nm) + os * __expf(om - nm);
            m = nm;
        }
        out[strip * 32 + tid] = m + __logf(s) - LOG_M;
    }
}

// ---------------------------------------------------------------------------
extern "C" void kernel_launch(void* const* d_in, const int* in_sizes, int n_in,
                              void* d_out, int out_size, void* d_ws, size_t ws_size,
                              hipStream_t stream) {
    (void)in_sizes; (void)n_in; (void)out_size; (void)ws_size;
    const float* samples = (const float*)d_in[0];
    const float* x       = (const float*)d_in[1];
    const float* stdv    = (const float*)d_in[2];
    float* out = (float*)d_out;
    char*  ws  = (char*)d_ws;

    // workspace layout (~39.4 MB)
    float*    inv_std  = (float*)(ws);                 // 800*4
    float*    logconst = (float*)(ws + 3200);
    float*    unorm    = (float*)(ws + 4096);          // 8192*4
    float*    vnorm    = (float*)(ws + 4096 + 32768);  // 4096*4
    _Float16* Apk      = (_Float16*)(ws + 65536);      // 26,214,400 B
    _Float16* Bpk      = (_Float16*)(ws + 65536 + 26214400); // 13,107,200 B

    k_std<<<1, 256, 0, stream>>>(stdv, inv_std, logconst);
    k_norm<<<NN, 256, 0, stream>>>(samples, inv_std, unorm);
    k_norm<<<MM, 256, 0, stream>>>(x, inv_std, vnorm);

    const int packA = (int)(((size_t)NTILES * KT * 512 + 255) / 256);
    const int packB = (int)(((size_t)KT * MTILES * 512 + 255) / 256);
    k_pack_a<<<packA, 256, 0, stream>>>(samples, inv_std, Apk);
    k_pack_b<<<packB, 256, 0, stream>>>(x, inv_std, Bpk);

    // dynamic LDS: A strip 102400 + 2 x 40960 B chunk buffers = 184320 B
    k_gemm_lse<<<NN / 32, 256, 184320, stream>>>(
        (const char*)Apk, (const char*)Bpk, unorm, vnorm, logconst, out);
}
